// GroupedQueryAttention_86990267613687
// MI455X (gfx1250) — compile-verified, matrix operand order checked by reading, not run on hardware
//
#include <hip/hip_runtime.h>

#define EMBED   1024
#define NHEADS  16
#define NGROUPS 8
#define HDIM    64
#define BATCH   4
#define SEQ     2048
#define TOKENS  (BATCH * SEQ)     // 8192
#define QDIM    (NGROUPS * HDIM)  // 512

typedef __attribute__((ext_vector_type(16))) _Float16 v16h;
typedef __attribute__((ext_vector_type(8)))  _Float16 v8h;
typedef __attribute__((ext_vector_type(4)))  _Float16 v4h;
typedef __attribute__((ext_vector_type(8)))  float    v8f;
typedef __attribute__((ext_vector_type(4)))  float    v4f;

static __device__ __forceinline__ v16h cat8(v8h lo, v8h hi) {
  return __builtin_shufflevector(lo, hi, 0,1,2,3,4,5,6,7,8,9,10,11,12,13,14,15);
}

static __device__ __forceinline__ v8f wmma_f16(v16h a, v16h b, v8f c) {
  // (neg_a, A, neg_b, B, c_mod, C, reuse_a, reuse_b)
  return __builtin_amdgcn_wmma_f32_16x16x32_f16(false, a, false, b, (short)0, c, false, false);
}

// Async global -> LDS copy, 16 bytes per lane (tracked by ASYNCcnt).
static __device__ __forceinline__ void async_load_b128(_Float16* ldsDst,
                                                       const _Float16* gSrc) {
  unsigned ldsOff = (unsigned)(unsigned long long)ldsDst;  // low 32b = LDS offset
  asm volatile("global_load_async_to_lds_b128 %0, %1, off"
               :: "v"(ldsOff), "v"(gSrc) : "memory");
}
static __device__ __forceinline__ void wait_asynccnt0() {
  asm volatile("s_wait_asynccnt 0" ::: "memory");
}

// ---------------------------------------------------------------------------
// 1) f32 -> f16 elementwise convert (vectorized by 4)
// ---------------------------------------------------------------------------
__global__ void cvt_f32_f16_kernel(const float* __restrict__ src,
                                   _Float16* __restrict__ dst, int n4) {
  int i = blockIdx.x * blockDim.x + threadIdx.x;
  if (i < n4) {
    v4f v = ((const v4f*)src)[i];
    ((v4h*)dst)[i] = __builtin_convertvector(v, v4h);
  }
}

// ---------------------------------------------------------------------------
// 2) W[K][N] f32 -> Wt[N][K] f16 (transpose + convert)
// ---------------------------------------------------------------------------
__global__ void transpose_cvt_kernel(const float* __restrict__ W,
                                     _Float16* __restrict__ Wt, int K, int N) {
  int idx = blockIdx.x * blockDim.x + threadIdx.x;
  if (idx < K * N) {
    int k = idx / N;
    int n = idx - k * N;
    Wt[(size_t)n * K + k] = (_Float16)W[idx];
  }
}

// ---------------------------------------------------------------------------
// 3) V[b*S+s][h*64+d] -> Vt[((b*H+h)*64+d)][s]  (per-head transpose)
// ---------------------------------------------------------------------------
__global__ void v_head_transpose_kernel(const _Float16* __restrict__ V,
                                        _Float16* __restrict__ Vt) {
  int idx = blockIdx.x * blockDim.x + threadIdx.x;
  if (idx < TOKENS * EMBED) {
    int d = idx & (HDIM - 1);
    int h = (idx >> 6) & (NHEADS - 1);
    int s = (idx >> 10) & (SEQ - 1);
    int b = idx >> 21;
    Vt[((size_t)((b * NHEADS + h) * HDIM) + d) * SEQ + s] = V[idx];
  }
}

// ---------------------------------------------------------------------------
// 4) NT GEMM: C[M][N] = A[M][K] * Bt[N][K]^T + bias[N]
//    128x128 block tile, 8 waves (4 along M x 2 along N), each wave 32x64.
//    Double-buffered LDS fed by GLOBAL_LOAD_ASYNC_TO_LDS_B128 (ASYNCcnt),
//    one barrier per 32-deep K-step; WMMA f32_16x16x32_f16.
// ---------------------------------------------------------------------------
#define BM 128
#define BN 128
#define KSTEP 32
#define LSTR 40  // LDS row stride in halves (32 + 8 pad, 80B = 5x16B aligned)

template <bool OUT_F32>
__global__ __launch_bounds__(256) void gemm_nt_kernel(
    const _Float16* __restrict__ A,   // [M][K]
    const _Float16* __restrict__ Bt,  // [N][K]
    const float*    __restrict__ bias,// [N]
    _Float16* __restrict__ Ch,        // f16 out (when !OUT_F32)
    float*    __restrict__ Cf,        // f32 out (when OUT_F32)
    int M, int N, int K) {
  __shared__ __attribute__((aligned(16))) _Float16 lA[2][BM * LSTR];
  __shared__ __attribute__((aligned(16))) _Float16 lB[2][BN * LSTR];

  const int nTilesN = N / BN;
  const int tileM = blockIdx.x / nTilesN;
  const int tileN = blockIdx.x % nTilesN;
  const int tid  = threadIdx.x;
  const int lane = tid & 31;
  const int wave = tid >> 5;
  const int wM = (wave & 3) * 32;  // 4 waves along M
  const int wN = (wave >> 2) * 64; // 2 waves along N
  const int half = lane >> 4;
  const int l16  = lane & 15;

  v8f acc[2][4] = {};

  const int c0 = tid * 2;  // 512 v8h chunks per 128x32 tile, 2 per thread
  const int r0  = (c0) >> 2,      kc0 = ((c0) & 3) * 8;
  const int r1  = (c0 + 1) >> 2,  kc1 = ((c0 + 1) & 3) * 8;

  auto issue = [&](int buf, int k0) {
    async_load_b128(&lA[buf][r0 * LSTR + kc0],
                    A + (size_t)(tileM * BM + r0) * K + k0 + kc0);
    async_load_b128(&lB[buf][r0 * LSTR + kc0],
                    Bt + (size_t)(tileN * BN + r0) * K + k0 + kc0);
    async_load_b128(&lA[buf][r1 * LSTR + kc1],
                    A + (size_t)(tileM * BM + r1) * K + k0 + kc1);
    async_load_b128(&lB[buf][r1 * LSTR + kc1],
                    Bt + (size_t)(tileN * BN + r1) * K + k0 + kc1);
  };

  issue(0, 0);
  const int nsteps = K / KSTEP;

  for (int s = 0; s < nsteps; ++s) {
    wait_asynccnt0();    // this wave's async stores to LDS done
    __syncthreads();     // everyone's tile is resident
    if (s + 1 < nsteps) issue((s + 1) & 1, (s + 1) * KSTEP);

    const _Float16* bufA = &lA[s & 1][0];
    const _Float16* bufB = &lB[s & 1][0];
    v16h afrag[2], bfrag[4];
#pragma unroll
    for (int mi = 0; mi < 2; ++mi) {
      const _Float16* p = bufA + (wM + mi * 16 + l16) * LSTR + half * 8;
      afrag[mi] = cat8(*(const v8h*)p, *(const v8h*)(p + 16));
    }
#pragma unroll
    for (int ni = 0; ni < 4; ++ni) {
      const _Float16* p = bufB + (wN + ni * 16 + l16) * LSTR + half * 8;
      bfrag[ni] = cat8(*(const v8h*)p, *(const v8h*)(p + 16));
    }
#pragma unroll
    for (int mi = 0; mi < 2; ++mi)
#pragma unroll
      for (int ni = 0; ni < 4; ++ni)
        acc[mi][ni] = wmma_f16(afrag[mi], bfrag[ni], acc[mi][ni]);
  }

  // C/D layout: lane l -> col = l16; rows = half*8 + vr
#pragma unroll
  for (int mi = 0; mi < 2; ++mi) {
#pragma unroll
    for (int ni = 0; ni < 4; ++ni) {
      int gn = tileN * BN + wN + ni * 16 + l16;
      float bv = bias[gn];
      int gm0 = tileM * BM + wM + mi * 16 + half * 8;
#pragma unroll
      for (int vr = 0; vr < 8; ++vr) {
        float v = acc[mi][ni][vr] + bv;
        size_t off = (size_t)(gm0 + vr) * N + gn;
        if constexpr (OUT_F32) Cf[off] = v;
        else                   Ch[off] = (_Float16)v;
      }
    }
  }
}

// ---------------------------------------------------------------------------
// 5) GQA flash attention, software-pipelined.
//    Grid: B * H * (S/64) blocks, 128 threads (4 waves, 16 q-rows each).
//    Per 32-key step: 4 WMMA (Q*K^T) on preloaded K-frags; V-frags and the
//    NEXT step's K-frags are loaded under the f32 online-softmax VALU work;
//    P goes through per-wave LDS (s_wait_dscnt) into A-layout; 4 WMMA (P*V).
// ---------------------------------------------------------------------------
#define KT 32

__global__ __launch_bounds__(128) void gqa_attention_kernel(
    const _Float16* __restrict__ Qh,  // [TOKENS][QDIM]
    const _Float16* __restrict__ Kh,  // [TOKENS][EMBED]
    const _Float16* __restrict__ Vt,  // [B*H*HDIM][SEQ]
    _Float16* __restrict__ Oh) {      // [TOKENS][EMBED]
  __shared__ __attribute__((aligned(16))) _Float16 lP[4][16 * KT];

  const int wave = threadIdx.x >> 5;
  const int lane = threadIdx.x & 31;
  const int half = lane >> 4;
  const int l16  = lane & 15;

  int bid = blockIdx.x;
  const int qBlocks = SEQ / 64;
  const int qb = bid % qBlocks; bid /= qBlocks;
  const int h  = bid % NHEADS;  bid /= NHEADS;
  const int b  = bid;
  const int g  = h >> 1;  // query group (2 heads per group)

  const int q0 = qb * 64 + wave * 16;
  const size_t tokQ = (size_t)b * SEQ + q0;

  // Q A-frags, pre-scaled by 64^-0.5 = 0.125 (exact in f16)
  v16h qfrag[2];
#pragma unroll
  for (int kd = 0; kd < 2; ++kd) {
    const _Float16* p = Qh + (tokQ + l16) * QDIM + g * HDIM + kd * 32 + half * 8;
    v16h q = cat8(*(const v8h*)p, *(const v8h*)(p + 16));
#pragma unroll
    for (int e = 0; e < 16; ++e) q[e] = q[e] * (_Float16)0.125f;
    qfrag[kd] = q;
  }

  v8f o[4] = {};
  float mrow[8], lrow[8];
#pragma unroll
  for (int vr = 0; vr < 8; ++vr) { mrow[vr] = -1e30f; lrow[vr] = 0.0f; }

  _Float16* Pbuf = &lP[wave][0];
  const size_t kRowBase = (size_t)b * SEQ;
  const size_t vhBase = (size_t)((b * NHEADS + h) * HDIM) * SEQ;

  auto loadK = [&](int s0, int ki, int kd) -> v16h {
    const _Float16* p = Kh + (kRowBase + s0 + ki * 16 + l16) * EMBED
                        + h * HDIM + kd * 32 + half * 8;
    return cat8(*(const v8h*)p, *(const v8h*)(p + 16));
  };

  // prologue: K-frags for step 0
  v16h kf[2][2];
#pragma unroll
  for (int ki = 0; ki < 2; ++ki)
#pragma unroll
    for (int kd = 0; kd < 2; ++kd) kf[ki][kd] = loadK(0, ki, kd);

  for (int s0 = 0; s0 < SEQ; s0 += KT) {
    // ---- scores = (Q*scale) K^T on preloaded frags
    v8f sc[2] = {};
#pragma unroll
    for (int ki = 0; ki < 2; ++ki)
#pragma unroll
      for (int kd = 0; kd < 2; ++kd)
        sc[ki] = wmma_f16(qfrag[kd], kf[ki][kd], sc[ki]);

    // ---- issue V-frag loads for this step (consumed after softmax)
    v16h vf[4];
#pragma unroll
    for (int ni = 0; ni < 4; ++ni) {
      const _Float16* p = Vt + vhBase + (size_t)(ni * 16 + l16) * SEQ + s0 + half * 8;
      vf[ni] = cat8(*(const v8h*)p, *(const v8h*)(p + 16));
    }
    // ---- preload next step's K-frags (hide under softmax VALU)
    const bool hasNext = (s0 + KT) < SEQ;
    v16h kfn[2][2];
    if (hasNext) {
#pragma unroll
      for (int ki = 0; ki < 2; ++ki)
#pragma unroll
        for (int kd = 0; kd < 2; ++kd) kfn[ki][kd] = loadK(s0 + KT, ki, kd);
    }

    // ---- online softmax (rows live at (half, vr); cols across 16 lanes)
    float alpha[8];
#pragma unroll
    for (int vr = 0; vr < 8; ++vr) {
      float mx = fmaxf(sc[0][vr], sc[1][vr]);
#pragma unroll
      for (int m = 1; m < 16; m <<= 1) mx = fmaxf(mx, __shfl_xor(mx, m, 32));
      float mnew = fmaxf(mrow[vr], mx);
      alpha[vr] = __expf(mrow[vr] - mnew);
      float p0 = __expf(sc[0][vr] - mnew);
      float p1 = __expf(sc[1][vr] - mnew);
      sc[0][vr] = p0;
      sc[1][vr] = p1;
      float rsum = p0 + p1;
#pragma unroll
      for (int m = 1; m < 16; m <<= 1) rsum += __shfl_xor(rsum, m, 32);
      lrow[vr] = lrow[vr] * alpha[vr] + rsum;
      mrow[vr] = mnew;
    }
#pragma unroll
    for (int ni = 0; ni < 4; ++ni)
#pragma unroll
      for (int vr = 0; vr < 8; ++vr) o[ni][vr] *= alpha[vr];

    // ---- P: C-layout -> LDS [q][key] -> reload in A-layout
#pragma unroll
    for (int ki = 0; ki < 2; ++ki)
#pragma unroll
      for (int vr = 0; vr < 8; ++vr)
        Pbuf[(half * 8 + vr) * KT + ki * 16 + l16] = (_Float16)sc[ki][vr];
    asm volatile("s_wait_dscnt 0" ::: "memory");
    v16h pf;
    {
      const _Float16* p = &Pbuf[l16 * KT + half * 8];
      pf = cat8(*(const v8h*)p, *(const v8h*)(p + 16));
    }
    // ---- O += P * V
#pragma unroll
    for (int ni = 0; ni < 4; ++ni) o[ni] = wmma_f16(pf, vf[ni], o[ni]);

    if (hasNext) {
#pragma unroll
      for (int ki = 0; ki < 2; ++ki)
#pragma unroll
        for (int kd = 0; kd < 2; ++kd) kf[ki][kd] = kfn[ki][kd];
    }
  }

  // ---- normalize and store (f16, [token][h*64+d])
#pragma unroll
  for (int ni = 0; ni < 4; ++ni) {
#pragma unroll
    for (int vr = 0; vr < 8; ++vr) {
      float v = o[ni][vr] / lrow[vr];
      int q = q0 + half * 8 + vr;
      Oh[((size_t)b * SEQ + q) * EMBED + h * HDIM + ni * 16 + l16] = (_Float16)v;
    }
  }
}

// ---------------------------------------------------------------------------
// Launch
// ---------------------------------------------------------------------------
extern "C" void kernel_launch(void* const* d_in, const int* in_sizes, int n_in,
                              void* d_out, int out_size, void* d_ws, size_t ws_size,
                              hipStream_t stream) {
  const float* x  = (const float*)d_in[0];
  const float* Wq = (const float*)d_in[1];
  const float* bq = (const float*)d_in[2];
  const float* Wk = (const float*)d_in[3];
  const float* bk = (const float*)d_in[4];
  const float* Wv = (const float*)d_in[5];
  const float* bv = (const float*)d_in[6];
  const float* Wo = (const float*)d_in[7];
  const float* bo = (const float*)d_in[8];
  float* out = (float*)d_out;

  char* ws = (char*)d_ws;
  const size_t XH_OFF  = 0;                                     // 16 MB (Vt later)
  const size_t QH_OFF  = XH_OFF  + (size_t)TOKENS * EMBED * 2;  // 8 MB
  const size_t KH_OFF  = QH_OFF  + (size_t)TOKENS * QDIM  * 2;  // 16 MB
  const size_t VH_OFF  = KH_OFF  + (size_t)TOKENS * EMBED * 2;  // 16 MB
  const size_t OH_OFF  = VH_OFF  + (size_t)TOKENS * EMBED * 2;  // 16 MB
  const size_t WQT_OFF = OH_OFF  + (size_t)TOKENS * EMBED * 2;  // 1 MB
  const size_t WKT_OFF = WQT_OFF + (size_t)EMBED * QDIM  * 2;   // 2 MB
  const size_t WVT_OFF = WKT_OFF + (size_t)EMBED * EMBED * 2;   // 2 MB
  const size_t WOT_OFF = WVT_OFF + (size_t)EMBED * EMBED * 2;   // 2 MB

  _Float16* xh  = (_Float16*)(ws + XH_OFF);
  _Float16* vt  = (_Float16*)(ws + XH_OFF);  // alias: x dead after projections
  _Float16* qh  = (_Float16*)(ws + QH_OFF);
  _Float16* kh  = (_Float16*)(ws + KH_OFF);
  _Float16* vh  = (_Float16*)(ws + VH_OFF);
  _Float16* oh  = (_Float16*)(ws + OH_OFF);
  _Float16* wqt = (_Float16*)(ws + WQT_OFF);
  _Float16* wkt = (_Float16*)(ws + WKT_OFF);
  _Float16* wvt = (_Float16*)(ws + WVT_OFF);
  _Float16* wot = (_Float16*)(ws + WOT_OFF);

  {
    int n4 = TOKENS * EMBED / 4;
    cvt_f32_f16_kernel<<<(n4 + 255) / 256, 256, 0, stream>>>(x, xh, n4);
  }
  transpose_cvt_kernel<<<(EMBED * QDIM  + 255) / 256, 256, 0, stream>>>(Wq, wqt, EMBED, QDIM);
  transpose_cvt_kernel<<<(EMBED * EMBED + 255) / 256, 256, 0, stream>>>(Wk, wkt, EMBED, EMBED);
  transpose_cvt_kernel<<<(EMBED * EMBED + 255) / 256, 256, 0, stream>>>(Wv, wvt, EMBED, EMBED);
  transpose_cvt_kernel<<<(EMBED * EMBED + 255) / 256, 256, 0, stream>>>(Wo, wot, EMBED, EMBED);

  gemm_nt_kernel<false><<<(TOKENS / BM) * (QDIM / BN), 256, 0, stream>>>(
      xh, wqt, bq, qh, nullptr, TOKENS, QDIM, EMBED);
  gemm_nt_kernel<false><<<(TOKENS / BM) * (EMBED / BN), 256, 0, stream>>>(
      xh, wkt, bk, kh, nullptr, TOKENS, EMBED, EMBED);
  gemm_nt_kernel<false><<<(TOKENS / BM) * (EMBED / BN), 256, 0, stream>>>(
      xh, wvt, bv, vh, nullptr, TOKENS, EMBED, EMBED);

  v_head_transpose_kernel<<<(TOKENS * EMBED + 255) / 256, 256, 0, stream>>>(vh, vt);

  gqa_attention_kernel<<<BATCH * NHEADS * (SEQ / 64), 128, 0, stream>>>(qh, kh, vt, oh);

  gemm_nt_kernel<true><<<(TOKENS / BM) * (EMBED / BN), 256, 0, stream>>>(
      oh, wot, bo, nullptr, out, TOKENS, EMBED, EMBED);

  (void)in_sizes; (void)n_in; (void)out_size; (void)ws_size;
}
